// InstanceMatcher_51582557225228
// MI455X (gfx1250) — matrix operand order repaired and checked by reading, not compiled, not run on hardware
//
#include <hip/hip_runtime.h>
#include <hip/hip_bf16.h>
#include <math.h>

// Problem constants (fixed by the reference).
constexpr int BATCH = 4;
constexpr int NDIM  = 4096;   // H*W
constexpr int CDIM  = 1024;

typedef __attribute__((ext_vector_type(16))) __bf16         bf16x16;
typedef __attribute__((ext_vector_type(8)))  float          f32x8;
typedef __attribute__((ext_vector_type(4)))  unsigned int   u32x4;
typedef __attribute__((ext_vector_type(8)))  unsigned int   u32x8;
typedef __attribute__((ext_vector_type(4)))  unsigned short u16x4;

__device__ __forceinline__ unsigned short f2bf(float f) {
    unsigned u = __float_as_uint(f);
    u += 0x7FFFu + ((u >> 16) & 1u);   // round-to-nearest-even
    return (unsigned short)(u >> 16);
}

// ---------------------------------------------------------------------------
// Kernel 1: L2-normalize each C-dim row, emit bf16 row-major matrix.
// ---------------------------------------------------------------------------
__global__ void normalize_kernel(const float* __restrict__ x,
                                 unsigned short* __restrict__ fnb) {
    const int lane = threadIdx.x & 31;
    const int wid  = threadIdx.x >> 5;
    const long row = (long)blockIdx.x * (blockDim.x >> 5) + wid;  // 0..B*N-1

    const float4* xr = (const float4*)(x + row * (long)CDIM);
    float4 vals[8];
    float ss = 0.f;
#pragma unroll
    for (int it = 0; it < 8; ++it) {
        float4 v = xr[it * 32 + lane];
        vals[it] = v;
        ss += v.x * v.x + v.y * v.y + v.z * v.z + v.w * v.w;
    }
#pragma unroll
    for (int off = 16; off > 0; off >>= 1)
        ss += __shfl_xor(ss, off, 32);
    const float inv = 1.0f / fmaxf(sqrtf(ss), 1e-12f);

    u16x4* orow = (u16x4*)(fnb + row * (long)CDIM);
#pragma unroll
    for (int it = 0; it < 8; ++it) {
        u16x4 o;
        o[0] = f2bf(vals[it].x * inv);
        o[1] = f2bf(vals[it].y * inv);
        o[2] = f2bf(vals[it].z * inv);
        o[3] = f2bf(vals[it].w * inv);
        orow[it * 32 + lane] = o;
    }
}

// ---------------------------------------------------------------------------
// Kernel 2: sim = fn . fn^T with LDS-staged bf16 WMMA.
//
// Workgroup (256 thr / 8 waves) owns a 128x128 output tile. Per K-chunk of 32:
//   * async-copy A(128x32) and B(128x32) bf16 panels global->LDS with
//     global_load_async_to_lds_b128 (coalesced 16B/lane, ASYNCcnt-tracked),
//   * s_wait_asynccnt 0 + barrier,
//   * each wave reads its operands from LDS (stride-80 rows: 16B-aligned,
//     bank-conflict-free ds_load_b128) and issues 8 v_wmma_f32_16x16x32_bf16
//     (a 64x32 sub-tile: waves arranged 2x4 over the 128x128 block).
// Arithmetic intensity vs global memory: ~67 FLOP/B -> compute-bound.
//
// Operand lane layouts used by this code (bf16, contiguous k, same for A and B):
//   A 16x32: lanes 0-15 rows, bytes [0,32); lanes 16-31 bytes [32,64)
//   B 32x16: lanes 0-15 cols, bytes [0,32); lanes 16-31 bytes [32,64)
// ---------------------------------------------------------------------------
constexpr int LDS_STRIDE = 80;   // 64 data bytes + 16 pad: aligned, conflict-free

__global__ void gemm_kernel(const unsigned short* __restrict__ fnb,
                            float* __restrict__ sim) {
    __shared__ unsigned char smemA[128 * LDS_STRIDE];
    __shared__ unsigned char smemB[128 * LDS_STRIDE];

    const int tid  = threadIdx.x;
    const int lane = tid & 31;
    const int wid  = tid >> 5;
    const int half = lane >> 4;       // 0: low 16 lanes, 1: high 16 lanes
    const int l15  = lane & 15;
    const int wr   = wid >> 2;        // wave row   0..1  (64 rows each)
    const int wc   = wid & 3;         // wave col   0..3  (32 cols each)

    // Block -> (batch, 128x128 tile)
    const int tilesPerDim = NDIM / 128;                    // 32
    const int b   = blockIdx.x / (tilesPerDim * tilesPerDim);
    const int rem = blockIdx.x % (tilesPerDim * tilesPerDim);
    const int tm  = (rem / tilesPerDim) * 128;
    const int tn  = (rem % tilesPerDim) * 128;

    const unsigned short* base = fnb + (size_t)b * NDIM * CDIM;

    // Staging assignment: thread t copies 32B of one row of A and one row of B.
    const int srow = tid >> 1;                 // 0..127
    const int soff = (tid & 1) * 32;           // byte offset within 64B row
    const char* gA = (const char*)(base + (size_t)(tm + srow) * CDIM) + soff;
    const char* gB = (const char*)(base + (size_t)(tn + srow) * CDIM) + soff;
    const unsigned ldsA = (unsigned)(uintptr_t)(&smemA[srow * LDS_STRIDE + soff]);
    const unsigned ldsB = (unsigned)(uintptr_t)(&smemB[srow * LDS_STRIDE + soff]);

    // Per-wave LDS read bases (byte addresses), contiguous-k operand layouts (A and B alike).
    const unsigned char* aBase = &smemA[(wr * 64 + l15) * LDS_STRIDE + half * 32];
    const unsigned char* bBase = &smemB[(wc * 32 + l15) * LDS_STRIDE + half * 32];

    f32x8 acc[4][2] = {};

    for (int kc = 0; kc < CDIM; kc += 32) {
        const int kb = kc * 2;   // byte advance along K in global rows

        // --- async stage: 4 x 16B per thread, ASYNCcnt-tracked ---
        asm volatile("global_load_async_to_lds_b128 %0, %1, off"
                     :: "v"(ldsA), "v"(gA + kb) : "memory");
        asm volatile("global_load_async_to_lds_b128 %0, %1, off offset:16"
                     :: "v"(ldsA), "v"(gA + kb) : "memory");
        asm volatile("global_load_async_to_lds_b128 %0, %1, off"
                     :: "v"(ldsB), "v"(gB + kb) : "memory");
        asm volatile("global_load_async_to_lds_b128 %0, %1, off offset:16"
                     :: "v"(ldsB), "v"(gB + kb) : "memory");
        asm volatile("s_wait_asynccnt 0" ::: "memory");
        __syncthreads();

        // --- B operands (shared across the 4 A sub-tiles) ---
        bf16x16 Bv[2];
#pragma unroll
        for (int j = 0; j < 2; ++j) {
            const u32x4* pb = (const u32x4*)(bBase + j * 16 * LDS_STRIDE);
            u32x4 lo = pb[0], hi = pb[1];
            u32x8 ub = {lo[0], lo[1], lo[2], lo[3], hi[0], hi[1], hi[2], hi[3]};
            Bv[j] = __builtin_bit_cast(bf16x16, ub);
        }
        // --- A operands + WMMA ---
#pragma unroll
        for (int i = 0; i < 4; ++i) {
            const u32x4* pa = (const u32x4*)(aBase + i * 16 * LDS_STRIDE);
            u32x4 lo = pa[0], hi = pa[1];          // +0B and +16B
            u32x8 ua = {lo[0], lo[1], lo[2], lo[3], hi[0], hi[1], hi[2], hi[3]};
            bf16x16 Av = __builtin_bit_cast(bf16x16, ua);
            acc[i][0] = __builtin_amdgcn_wmma_f32_16x16x32_bf16(
                false, Av, false, Bv[0], (short)0, acc[i][0], false, false);
            acc[i][1] = __builtin_amdgcn_wmma_f32_16x16x32_bf16(
                false, Av, false, Bv[1], (short)0, acc[i][1], false, false);
        }
        __syncthreads();   // LDS consumed before next chunk overwrites it
    }

    // C/D layout (ISA 7.12.2): lanes 0-15 -> M = r, lanes 16-31 -> M = r + 8.
    float* out = sim + (size_t)b * NDIM * NDIM;
#pragma unroll
    for (int i = 0; i < 4; ++i) {
        const int rowBase = tm + wr * 64 + i * 16 + (half ? 8 : 0);
#pragma unroll
        for (int j = 0; j < 2; ++j) {
            const int col = tn + wc * 32 + j * 16 + l15;
#pragma unroll
            for (int r = 0; r < 8; ++r)
                out[(size_t)(rowBase + r) * NDIM + col] = acc[i][j][r];
        }
    }
}

// ---------------------------------------------------------------------------
// Kernel 3: per-row max of sim over non-ref columns. One wave per row.
// ---------------------------------------------------------------------------
__global__ void rowmax_kernel(const float* __restrict__ sim,
                              const int* __restrict__ masks,
                              float* __restrict__ maxvals) {
    const int lane = threadIdx.x & 31;
    const int wid  = threadIdx.x >> 5;
    const long row = (long)blockIdx.x * (blockDim.x >> 5) + wid;  // 0..B*N-1
    const int  b   = (int)(row / NDIM);

    const float* s = sim + row * (long)NDIM;
    const int*   m = masks + (size_t)b * NDIM;

    float mx = -INFINITY;
    for (int j = lane; j < NDIM; j += 32)
        if (!m[j]) mx = fmaxf(mx, s[j]);
#pragma unroll
    for (int off = 16; off > 0; off >>= 1)
        mx = fmaxf(mx, __shfl_xor(mx, off, 32));
    if (lane == 0) maxvals[row] = mx;
}

// ---------------------------------------------------------------------------
// Kernel 4: exact lower-median over ref rows via 32-step radix select on
// order-preserving uint keys. One block per batch item.
// ---------------------------------------------------------------------------
__global__ void median_kernel(const float* __restrict__ maxvals,
                              const int* __restrict__ masks,
                              float* __restrict__ threshOut) {
    const int b   = blockIdx.x;
    const int tid = threadIdx.x;
    const float* mv = maxvals + (size_t)b * NDIM;
    const int*   m  = masks   + (size_t)b * NDIM;

    __shared__ int s_red[256];

    int local = 0;
    for (int i = tid; i < NDIM; i += blockDim.x) local += (m[i] != 0);
    s_red[tid] = local;
    __syncthreads();
    for (int s = 128; s > 0; s >>= 1) {
        if (tid < s) s_red[tid] += s_red[tid + s];
        __syncthreads();
    }
    const int k = s_red[0];
    const int target = (k - 1) / 2 + 1;   // rank (1-based) of lower median
    __syncthreads();

    unsigned res = 0;
    for (int bit = 31; bit >= 0; --bit) {
        const unsigned cand = res | (1u << bit);
        int c = 0;
        for (int i = tid; i < NDIM; i += blockDim.x) {
            if (m[i]) {
                const unsigned fb  = __float_as_uint(mv[i]);
                const unsigned key = (fb & 0x80000000u) ? ~fb : (fb | 0x80000000u);
                c += (key < cand);
            }
        }
        s_red[tid] = c;
        __syncthreads();
        for (int s = 128; s > 0; s >>= 1) {
            if (tid < s) s_red[tid] += s_red[tid + s];
            __syncthreads();
        }
        if (s_red[0] < target) res = cand;
        __syncthreads();
    }
    if (tid == 0) {
        const unsigned fb = (res & 0x80000000u) ? (res ^ 0x80000000u) : ~res;
        threshOut[b] = __uint_as_float(fb);
    }
}

// ---------------------------------------------------------------------------
// Kernel 5: match = ref(i) & !ref(j) & (sim > thresh[b]), emitted as f32 0/1.
// ---------------------------------------------------------------------------
__global__ void match_kernel(const float* __restrict__ sim,
                             const int* __restrict__ masks,
                             const float* __restrict__ thresh,
                             float* __restrict__ match) {
    const size_t idx = (size_t)blockIdx.x * blockDim.x + threadIdx.x;
    const size_t nn  = (size_t)NDIM * NDIM;
    const int b = (int)(idx / nn);
    const size_t r = idx % nn;
    const int i = (int)(r / NDIM);
    const int j = (int)(r % NDIM);
    const int* m = masks + (size_t)b * NDIM;
    match[idx] = (m[i] && !m[j] && (sim[idx] > thresh[b])) ? 1.0f : 0.0f;
}

// ---------------------------------------------------------------------------
// Host launcher. Scratch (bf16 fn: 32 MB, row maxima: 64 KB) lives inside the
// match region of d_out, which is fully overwritten by the final kernel, so
// we never depend on ws_size. Every byte of d_out is rewritten each call.
// ---------------------------------------------------------------------------
extern "C" void kernel_launch(void* const* d_in, const int* in_sizes, int n_in,
                              void* d_out, int out_size, void* d_ws, size_t ws_size,
                              hipStream_t stream) {
    const float* emb   = (const float*)d_in[0];
    const int*   masks = (const int*)d_in[1];

    float* out = (float*)d_out;
    const size_t NN = (size_t)NDIM * NDIM;
    float* matchOut  = out;
    float* simOut    = out + (size_t)BATCH * NN;
    float* threshOut = out + 2ull * BATCH * NN;

    unsigned short* fnb     = (unsigned short*)matchOut;                 // 32 MB scratch
    float*          maxvals = (float*)((char*)matchOut + (64ull << 20)); // 64 KB scratch

    // 1) normalize + bf16 convert: B*N rows, 8 waves/block -> exact grid.
    normalize_kernel<<<(BATCH * NDIM) / 8, 256, 0, stream>>>(emb, fnb);

    // 2) LDS-staged WMMA GEMM: B * 32 * 32 workgroups of 128x128 tiles.
    {
        const int blocks = BATCH * (NDIM / 128) * (NDIM / 128);  // 4096
        gemm_kernel<<<blocks, 256, 0, stream>>>(fnb, simOut);
    }

    // 3) per-row max over non-ref columns.
    rowmax_kernel<<<(BATCH * NDIM) / 8, 256, 0, stream>>>(simOut, masks, maxvals);

    // 4) per-batch lower median over ref rows.
    median_kernel<<<BATCH, 256, 0, stream>>>(maxvals, masks, threshOut);

    // 5) thresholded match map (overwrites the scratch region last).
    {
        const size_t total = (size_t)BATCH * NN;       // 67,108,864
        match_kernel<<<(int)(total / 256), 256, 0, stream>>>(simOut, masks, threshOut, matchOut);
    }
}
